// ReputationAwareNetwork_49890340110406
// MI455X (gfx1250) — matrix operand order repaired and checked by reading, not compiled, not run on hardware
//
#include <hip/hip_runtime.h>
#include <math.h>

// ---- problem constants (match reference) ----
#define DDIM   1024
#define HDIM   64
#define LDIM   32
#define NNEIGH 64
#define M1     128                       // 2*H
#define IN_DIM (DDIM*DDIM + HDIM)        // encoder input length
#define KBLOCK 2048                      // K-chunk per block for enc1
#define TILES_PER_WAVE 4                 // M-tiles per wave for dec3

typedef __attribute__((ext_vector_type(16))) _Float16 v16h;
typedef __attribute__((ext_vector_type(8)))  float    v8f;
typedef __attribute__((ext_vector_type(4)))  int      v4i;

// ---- gfx1250 async global->LDS path (guarded; falls back to VGPR bounce) ----
#if defined(__has_builtin)
#  if __has_builtin(__builtin_amdgcn_global_load_async_to_lds_b128)
#    define HAVE_ASYNC_LDS 1
#  else
#    define HAVE_ASYNC_LDS 0
#  endif
#else
#  define HAVE_ASYNC_LDS 0
#endif

#if HAVE_ASYNC_LDS
// builtin signature (from compiler diagnostic): param0 = v4i in addrspace(1)
typedef __attribute__((address_space(1))) v4i g1v4i;
typedef __attribute__((address_space(3))) v4i l3v4i;

__device__ __forceinline__ void async_b128_to_lds(const float* g, float* l) {
    // moves 16 bytes per active lane: LDS[l] = MEM[g]
    __builtin_amdgcn_global_load_async_to_lds_b128((g1v4i*)g, (l3v4i*)l, 0, 0);
}
__device__ __forceinline__ void wait_async0() {
#if __has_builtin(__builtin_amdgcn_s_wait_asynccnt)
    __builtin_amdgcn_s_wait_asynccnt(0);
#else
    asm volatile("s_wait_asynccnt 0" ::: "memory");
#endif
}
#endif

__device__ __forceinline__ float sigmoidf_(float x) {
    return 1.0f / (1.0f + __expf(-x));
}

// Pack a 16x32 f16 A-tile fragment for one lane.
// p points at W[row, kbase + hi*16]; we need K = {hi*16..hi*16+15}.
__device__ __forceinline__ v16h pack_a16(const float* __restrict__ p) {
    v16h a;
#pragma unroll
    for (int i = 0; i < 8; ++i) a[i]     = (_Float16)p[i];
#pragma unroll
    for (int i = 0; i < 8; ++i) a[8 + i] = (_Float16)p[8 + i];
    return a;
}

// ---------------------------------------------------------------------------
// K1: neighbor attention (tiny). Also zeroes the enc1 accumulator in ws.
// ws layout (floats): [0:64) attention_sum, [64:192) enc1 acc, [192:320) d2
// ---------------------------------------------------------------------------
__global__ void attn_kernel(const float* __restrict__ R_Di,
                            const float* __restrict__ Nbr,
                            const float* __restrict__ Wq,
                            const float* __restrict__ Wk,
                            const float* __restrict__ Wv,
                            float* __restrict__ ws) {
    __shared__ float Ksh[NNEIGH * HDIM];
    __shared__ float Vsh[NNEIGH * HDIM];
    __shared__ float qsh[HDIM];
    __shared__ float ash[NNEIGH];
    const int t = threadIdx.x;                 // 256 threads

    if (t < M1) ws[64 + t] = 0.0f;             // zero enc1 accumulator each call

    if (t < HDIM) {                            // q = W_q @ R_Di
        const float* w = Wq + t * DDIM;
        float s = 0.0f;
        for (int d = 0; d < DDIM; ++d) s = fmaf(w[d], R_Di[d], s);
        qsh[t] = s;
    }
    for (int idx = t; idx < NNEIGH * HDIM; idx += 256) {   // K, V
        const int n = idx >> 6, h = idx & 63;
        const float* x  = Nbr + n * DDIM;
        const float* wk = Wk + h * DDIM;
        const float* wv = Wv + h * DDIM;
        float sk = 0.0f, sv = 0.0f;
        for (int d = 0; d < DDIM; ++d) {
            const float xv = x[d];
            sk = fmaf(xv, wk[d], sk);
            sv = fmaf(xv, wv[d], sv);
        }
        Ksh[idx] = sk;
        Vsh[idx] = sv;
    }
    __syncthreads();
    if (t < NNEIGH) {                          // attn = clip(K@q / 8, +-10)
        float s = 0.0f;
        for (int h = 0; h < HDIM; ++h) s = fmaf(Ksh[t * HDIM + h], qsh[h], s);
        s *= 0.125f;
        ash[t] = fminf(10.0f, fmaxf(-10.0f, s));
    }
    __syncthreads();
    if (t < HDIM) {                            // attention_sum = attn @ V
        float s = 0.0f;
        for (int n = 0; n < NNEIGH; ++n) s = fmaf(ash[n], Vsh[n * HDIM + t], s);
        ws[t] = s;
    }
}

// ---------------------------------------------------------------------------
// K2: enc layer 1 over the R portion of `combined` (cols 64..IN_DIM) using
// WMMA f32_16x16x32_f16. Each block handles a KBLOCK slice of K for all 128
// output rows (8 waves x 16-row tiles); partials reduced in LDS then one
// atomicAdd per row. The shared R chunk is staged via the gfx1250
// async global->LDS engine when available.
// ---------------------------------------------------------------------------
__global__ void enc1_wmma_kernel(const float* __restrict__ Rflat,
                                 const float* __restrict__ W1,
                                 float* __restrict__ ws) {
    __shared__ __attribute__((aligned(64))) float    xs32[KBLOCK];
    __shared__ __attribute__((aligned(64))) _Float16 xs[KBLOCK];
    __shared__ float part[M1];
    const int t    = threadIdx.x;              // 256 threads, 8 waves
    const int lane = t & 31;
    const int wv   = t >> 5;
    const long kg0 = (long)blockIdx.x * KBLOCK;

#if HAVE_ASYNC_LDS
    // stage the R chunk into LDS via the async DMA path (16B per lane per op)
#pragma unroll
    for (int i = 0; i < KBLOCK / 4; i += 256) {
        const int e = (i + t) * 4;             // float4 index -> element
        async_b128_to_lds(Rflat + kg0 + e, xs32 + e);
    }
    wait_async0();
    __syncthreads();
    for (int i = t; i < KBLOCK; i += 256) xs[i] = (_Float16)xs32[i];
#else
    for (int i = t; i < KBLOCK; i += 256) xs[i] = (_Float16)Rflat[kg0 + i];
#endif
    __syncthreads();

    const int r    = lane & 15;
    const int hi   = lane >> 4;                // 0: lanes 0-15, 1: lanes 16-31
    const int row0 = wv * 16;
    const float* arow = W1 + (long)(row0 + r) * IN_DIM + 64 + kg0 + hi * 16;

    v8f c = {};
    for (int ks = 0; ks < KBLOCK; ks += 32) {
        __builtin_prefetch(arow + ks + 512, 0, 0);     // global_prefetch_b8
        v16h a = pack_a16(arow + ks);
        v16h b = *(const v16h*)(xs + ks + hi * 16);    // broadcast vector tile
        c = __builtin_amdgcn_wmma_f32_16x16x32_f16(
                false, a, false, b, (short)0, c, false, false);
    }

    // D layout: VGPR v, lanes0-15 -> M=v, lanes16-31 -> M=v+8 (column N=lane%16)
    if ((lane & 15) == 0) {
        const int mbase = row0 + hi * 8;
#pragma unroll
        for (int v = 0; v < 8; ++v) part[mbase + v] = c[v];
    }
    __syncthreads();
    if (t < M1) atomicAdd(&ws[64 + t], part[t]);
}

// ---------------------------------------------------------------------------
// K3: everything small between the two big layers (single 128-thread block):
// finish enc1 (attention_sum columns + bias + relu), enc2, VAE heads,
// reparameterization, dec1, dec2. Writes C_i/mean/logvar to out tail and the
// 128-long decoder activation d2 to ws.
// ---------------------------------------------------------------------------
__global__ void mid_kernel(const float* __restrict__ W1, const float* __restrict__ b1,
                           const float* __restrict__ W2, const float* __restrict__ b2,
                           const float* __restrict__ mw, const float* __restrict__ mb,
                           const float* __restrict__ vw, const float* __restrict__ vb,
                           const float* __restrict__ dw1, const float* __restrict__ db1,
                           const float* __restrict__ dw2, const float* __restrict__ db2,
                           const float* __restrict__ eps,
                           float* __restrict__ ws, float* __restrict__ out_tail) {
    __shared__ float h1[M1], Hi[HDIM], Ci[LDIM], d1[HDIM];
    const int t = threadIdx.x;                 // 128 threads

    if (t < M1) {                              // h = relu(W1@combined + b1)
        float s = ws[64 + t];                  // partial from R columns
        const float* w = W1 + (long)t * IN_DIM;
        for (int j = 0; j < HDIM; ++j) s = fmaf(w[j], ws[j], s);
        h1[t] = fmaxf(s + b1[t], 0.0f);
    }
    __syncthreads();
    if (t < HDIM) {                            // H_i = relu(W2@h + b2)
        float s = b2[t];
        const float* w = W2 + t * M1;
        for (int j = 0; j < M1; ++j) s = fmaf(w[j], h1[j], s);
        Hi[t] = fmaxf(s, 0.0f);
    }
    __syncthreads();
    if (t < LDIM) {                            // heads + reparameterization
        float sm = mb[t], sv = vb[t];
        for (int j = 0; j < HDIM; ++j) {
            sm = fmaf(mw[t * HDIM + j], Hi[j], sm);
            sv = fmaf(vw[t * HDIM + j], Hi[j], sv);
        }
        float lv = tanhf(sv);
        lv = fminf(4.0f, fmaxf(-4.0f, lv));
        const float sd = sqrtf(__expf(lv) + 1e-8f);
        const float ci = sm + eps[t] * sd;
        Ci[t] = ci;
        out_tail[t]      = ci;                 // C_i
        out_tail[32 + t] = sm;                 // C_i_mean
        out_tail[64 + t] = lv;                 // C_i_logvar
    }
    __syncthreads();
    if (t < HDIM) {                            // d1 = relu(dec_w1@C_i + b)
        float s = db1[t];
        for (int j = 0; j < LDIM; ++j) s = fmaf(dw1[t * LDIM + j], Ci[j], s);
        d1[t] = fmaxf(s, 0.0f);
    }
    __syncthreads();
    if (t < M1) {                              // d2 = relu(dec_w2@d1 + b)
        float s = db2[t];
        for (int j = 0; j < HDIM; ++j) s = fmaf(dw2[t * HDIM + j], d1[j], s);
        ws[192 + t] = fmaxf(s, 0.0f);
    }
}

// ---------------------------------------------------------------------------
// K4: dec layer 3: sigmoid(dec_w3 @ d2 + b3), 1M outputs, K=128, via WMMA.
// One 16-row tile per wave-iteration; B operands (d2 broadcast) staged once
// per block in LDS (async path when available).
// ---------------------------------------------------------------------------
__global__ void dec3_wmma_kernel(const float* __restrict__ W3,
                                 const float* __restrict__ b3,
                                 const float* __restrict__ ws,
                                 float* __restrict__ out) {
    __shared__ __attribute__((aligned(64))) float    dsh32[M1];
    __shared__ __attribute__((aligned(64))) _Float16 dsh[M1];
    const int t    = threadIdx.x;              // 256 threads, 8 waves
    const int lane = t & 31;
    const int wv   = t >> 5;

#if HAVE_ASYNC_LDS
    if (t < M1 / 4)                            // 32 lanes x 16B = 512B = d2
        async_b128_to_lds(ws + 192 + t * 4, dsh32 + t * 4);
    wait_async0();
    __syncthreads();
    if (t < M1) dsh[t] = (_Float16)dsh32[t];
#else
    if (t < M1) dsh[t] = (_Float16)ws[192 + t];
#endif
    __syncthreads();

    const int r  = lane & 15;
    const int hi = lane >> 4;
    const v16h b0  = *(const v16h*)(dsh +  0 + hi * 16);
    const v16h b1  = *(const v16h*)(dsh + 32 + hi * 16);
    const v16h b2  = *(const v16h*)(dsh + 64 + hi * 16);
    const v16h b3v = *(const v16h*)(dsh + 96 + hi * 16);

    const long tile0 = ((long)blockIdx.x * 8 + wv) * TILES_PER_WAVE;
    for (int tt = 0; tt < TILES_PER_WAVE; ++tt) {
        const long m0 = (tile0 + tt) * 16;
        const float* arow = W3 + (m0 + r) * 128 + hi * 16;
        __builtin_prefetch(arow + 2048, 0, 0);           // next tile's row
        v8f c = {};
        c = __builtin_amdgcn_wmma_f32_16x16x32_f16(false, pack_a16(arow +  0),
                false, b0,  (short)0, c, false, false);
        c = __builtin_amdgcn_wmma_f32_16x16x32_f16(false, pack_a16(arow + 32),
                false, b1,  (short)0, c, false, false);
        c = __builtin_amdgcn_wmma_f32_16x16x32_f16(false, pack_a16(arow + 64),
                false, b2,  (short)0, c, false, false);
        c = __builtin_amdgcn_wmma_f32_16x16x32_f16(false, pack_a16(arow + 96),
                false, b3v, (short)0, c, false, false);
        if ((lane & 15) == 0) {
            const long mb = m0 + hi * 8;
#pragma unroll
            for (int v = 0; v < 8; ++v)
                out[mb + v] = sigmoidf_(c[v] + b3[mb + v]);
        }
    }
}

// ---------------------------------------------------------------------------
extern "C" void kernel_launch(void* const* d_in, const int* in_sizes, int n_in,
                              void* d_out, int out_size, void* d_ws, size_t ws_size,
                              hipStream_t stream) {
    const float* R      = (const float*)d_in[0];
    const float* R_Di   = (const float*)d_in[1];
    const float* Nbr    = (const float*)d_in[2];
    const float* eps    = (const float*)d_in[3];
    const float* Wq     = (const float*)d_in[4];
    const float* Wk     = (const float*)d_in[5];
    const float* Wv     = (const float*)d_in[6];
    const float* enc_w1 = (const float*)d_in[7];
    const float* enc_b1 = (const float*)d_in[8];
    const float* enc_w2 = (const float*)d_in[9];
    const float* enc_b2 = (const float*)d_in[10];
    const float* mean_w = (const float*)d_in[11];
    const float* mean_b = (const float*)d_in[12];
    const float* var_w  = (const float*)d_in[13];
    const float* var_b  = (const float*)d_in[14];
    const float* dec_w1 = (const float*)d_in[15];
    const float* dec_b1 = (const float*)d_in[16];
    const float* dec_w2 = (const float*)d_in[17];
    const float* dec_b2 = (const float*)d_in[18];
    const float* dec_w3 = (const float*)d_in[19];
    const float* dec_b3 = (const float*)d_in[20];

    float* out = (float*)d_out;
    float* ws  = (float*)d_ws;

    hipLaunchKernelGGL(attn_kernel, dim3(1), dim3(256), 0, stream,
                       R_Di, Nbr, Wq, Wk, Wv, ws);

    hipLaunchKernelGGL(enc1_wmma_kernel, dim3((DDIM * DDIM) / KBLOCK), dim3(256),
                       0, stream, R, enc_w1, ws);

    hipLaunchKernelGGL(mid_kernel, dim3(1), dim3(128), 0, stream,
                       enc_w1, enc_b1, enc_w2, enc_b2,
                       mean_w, mean_b, var_w, var_b,
                       dec_w1, dec_b1, dec_w2, dec_b2,
                       eps, ws, out + (long)DDIM * DDIM);

    // 65536 tiles / 4 per wave / 8 waves per block = 2048 blocks
    hipLaunchKernelGGL(dec3_wmma_kernel, dim3(2048), dim3(256), 0, stream,
                       dec_w3, dec_b3, ws, out);
}
